// Alignment_42563125903561
// MI455X (gfx1250) — matrix operand order repaired and checked by reading, not compiled, not run on hardware
//
#include <hip/hip_runtime.h>

#define BATCH 64
#define SEQ   512
#define HD    768
#define KPITCH 40   // KhT row pitch in halfs: 80B rows -> 16B-aligned b128 segments,
                    // dword stride 20 (gcd(20,64)=4): 16 distinct banks per half-wave

typedef __attribute__((ext_vector_type(16))) _Float16 v16h;
typedef __attribute__((ext_vector_type(8)))  float    v8f;

// One flash-attention pass: O = softmax_rows(Q K^T * temp  [masked]) @ V, V == K.
// dir 0: Q=a, KV=b  -> feature_a ; dir 1: Q=b, KV=a -> feature_b.
// Workgroup: 8 waves, one 16-row Q tile. j streamed in chunks of 32.
// KV chunk is read from global exactly ONCE per chunk (as S-phase B-operands);
// the same registers are then scattered transposed into LDS for the PV phase.
__global__ __launch_bounds__(256)
void Alignment_fa_cdna5_kernel(const float* __restrict__ A,
                               const float* __restrict__ Bm,
                               const int*   __restrict__ MA,
                               const int*   __restrict__ MB,
                               const float* __restrict__ TP,
                               float*       __restrict__ OUT)
{
    __shared__ _Float16 KhT[HD * KPITCH];   // KV chunk, transposed (h-major), 60KB
    __shared__ float    Sred[16 * 32];      // split-h partial S reduction
    __shared__ _Float16 Ph[16 * 32];        // exp(S - m), f16
    __shared__ float    mstate[16], lstate[16], cfac[16];

    const int tid  = threadIdx.x;
    const int lane = tid & 31;
    const int wid  = tid >> 5;      // 0..7
    const int lhi  = lane >> 4;     // 0/1 (half-wave)
    const int llo  = lane & 15;

    const int qtile = blockIdx.x;   // 0..31
    const int batch = blockIdx.y;   // 0..63
    const int dir   = blockIdx.z;   // 0/1

    const float* Q  = dir ? Bm : A;
    const float* KV = dir ? A  : Bm;
    const int*   QM = dir ? MB : MA;
    const int*   KM = dir ? MA : MB;
    float* outp = OUT + (size_t)dir * (size_t)BATCH * SEQ * HD;

    const float temp  = *TP;
    const int   qbase = qtile * 16;
    const float* Qb  = Q  + ((size_t)batch * SEQ + qbase) * HD;
    const float* KVb = KV + (size_t)batch * SEQ * HD;
    const int*   qm  = QM + batch * SEQ;
    const int*   km  = KM + batch * SEQ;

    const int wslice = wid * 96;    // this wave's 96-wide h slice

    // ---- Preload this wave's Q A-operands (3 h-chunks of K=32), f32 -> f16 ----
    // A 16x32 f16 layout: lane m=llo; elems k -> K(lhi*16 + k) (contiguous)
    v16h qa[3];
#pragma unroll
    for (int hc = 0; hc < 3; ++hc) {
        const float* p1 = Qb + (size_t)llo * HD + wslice + hc * 32 + lhi * 16;
        const float* p2 = p1 + 8;
#pragma unroll
        for (int k = 0; k < 8; ++k) {
            qa[hc][k]     = (_Float16)p1[k];
            qa[hc][8 + k] = (_Float16)p2[k];
        }
    }

    v8f Oacc[6];
#pragma unroll
    for (int t = 0; t < 6; ++t)
#pragma unroll
        for (int r = 0; r < 8; ++r) Oacc[t][r] = 0.0f;

    // initial zeroing of Sred + state
    Sred[tid]       = 0.0f;
    Sred[tid + 256] = 0.0f;
    if (tid < 16) { mstate[tid] = -INFINITY; lstate[tid] = 0.0f; }
    __syncthreads();

    for (int ch = 0; ch < 16; ++ch) {
        const int jbase = ch * 32;

        // ---- S-phase: partial S = Q_slice @ K_slice^T for this wave's h slice.
        //      Each B-operand (global, f32->f16) is also scattered transposed
        //      into KhT for the PV phase: single global read of KV per chunk. ----
#pragma unroll
        for (int jt = 0; jt < 2; ++jt) {
            v8f sacc;
#pragma unroll
            for (int r = 0; r < 8; ++r) sacc[r] = 0.0f;
            const int jloc = jt * 16 + llo;
#pragma unroll
            for (int hc = 0; hc < 3; ++hc) {
                // B 32x16: lane col j = llo; elems k -> K = lhi*16 + k (contiguous in h)
                const float* p = KVb + (size_t)(jbase + jloc) * HD
                                     + wslice + hc * 32 + lhi * 16;
                if ((hc == 0) && (ch + 1 < 16))
                    __builtin_prefetch(p + 32 * HD, 0, 0);   // next chunk, same lane rows
                v16h bk;
#pragma unroll
                for (int k = 0; k < 16; ++k) bk[k] = (_Float16)p[k];
                sacc = __builtin_amdgcn_wmma_f32_16x16x32_f16(
                           false, qa[hc], false, bk, (short)0, sacc, false, false);
                // transpose-scatter into KhT: elem k lives at (h = base+k, col = jloc)
                _Float16* q = &KhT[(wslice + hc * 32 + lhi * 16) * KPITCH + jloc];
#pragma unroll
                for (int k = 0; k < 16; ++k) q[k * KPITCH] = bk[k];
            }
            // C/D layout: VGPR r, element (M = r + 8*lhi, N = llo)
#pragma unroll
            for (int r = 0; r < 8; ++r)
                atomicAdd(&Sred[(r + 8 * lhi) * 32 + jt * 16 + llo], sacc[r]);
        }
        __syncthreads();

        // ---- online softmax stats, rows handled by threads 0..15 ----
        if (tid < 16) {
            const int i  = tid;
            const int qv = qm[qbase + i];
            const float mprev = mstate[i];
            float mmax = mprev;
            for (int j = 0; j < 32; ++j) {
                float s = (qv && km[jbase + j]) ? Sred[i * 32 + j] * temp : -1.0e7f;
                mmax = fmaxf(mmax, s);
            }
            const float c = __expf(mprev - mmax);
            float sum = lstate[i] * c;
            for (int j = 0; j < 32; ++j) {
                float s = (qv && km[jbase + j]) ? Sred[i * 32 + j] * temp : -1.0e7f;
                float p = __expf(s - mmax);
                sum += p;
                Ph[i * 32 + j] = (_Float16)p;
            }
            mstate[i] = mmax; lstate[i] = sum; cfac[i] = c;
        }
        __syncthreads();

        // ---- PV phase: rescale O, O += P @ V; also re-zero Sred for next chunk ----
        Sred[tid]       = 0.0f;
        Sred[tid + 256] = 0.0f;

        float cr[8];
#pragma unroll
        for (int r = 0; r < 8; ++r) cr[r] = cfac[r + 8 * lhi];
#pragma unroll
        for (int t = 0; t < 6; ++t)
#pragma unroll
            for (int r = 0; r < 8; ++r) Oacc[t][r] *= cr[r];

        // P A-operand 16x32: row llo; elems k -> j(lhi*16+k) (contiguous)
        v16h pa;
        {
            const _Float16* p1 = &Ph[llo * 32 + lhi * 16];
            const _Float16* p2 = p1 + 8;
#pragma unroll
            for (int k = 0; k < 8; ++k) { pa[k] = p1[k]; pa[8 + k] = p2[k]; }
        }
#pragma unroll
        for (int t = 0; t < 6; ++t) {
            // V B-operand 32x16: col h = llo (within tile); elems k -> j = lhi*16 + k
            // 80B row pitch -> both 32B segments per lane are 16B aligned (ds_read_b128)
            const _Float16* p = &KhT[(wslice + t * 16 + llo) * KPITCH + lhi * 16];
            v16h bv;
#pragma unroll
            for (int k = 0; k < 16; ++k) bv[k] = p[k];
            Oacc[t] = __builtin_amdgcn_wmma_f32_16x16x32_f16(
                          false, pa, false, bv, (short)0, Oacc[t], false, false);
        }
        __syncthreads();   // protects KhT/Sred for next chunk's S-phase
    }

    // ---- normalize by 1/l and write fp32 output ----
    float linv[8];
#pragma unroll
    for (int r = 0; r < 8; ++r) linv[r] = 1.0f / lstate[r + 8 * lhi];
#pragma unroll
    for (int t = 0; t < 6; ++t) {
#pragma unroll
        for (int r = 0; r < 8; ++r) {
            const size_t row = (size_t)batch * SEQ + qbase + r + 8 * lhi;
            outp[row * HD + wslice + t * 16 + llo] = Oacc[t][r] * linv[r];
        }
    }
}

extern "C" void kernel_launch(void* const* d_in, const int* in_sizes, int n_in,
                              void* d_out, int out_size, void* d_ws, size_t ws_size,
                              hipStream_t stream) {
    (void)in_sizes; (void)n_in; (void)out_size; (void)d_ws; (void)ws_size;
    const float* a  = (const float*)d_in[0];
    const float* b  = (const float*)d_in[1];
    const int*   ma = (const int*)d_in[2];
    const int*   mb = (const int*)d_in[3];
    const float* tp = (const float*)d_in[4];
    float* out = (float*)d_out;

    dim3 grid(SEQ / 16, BATCH, 2);   // 32 q-tiles x 64 batches x 2 directions
    dim3 block(256);                 // 8 wave32 waves
    Alignment_fa_cdna5_kernel<<<grid, block, 0, stream>>>(a, b, ma, mb, tp, out);
}
